// GAT_18279380812366
// MI455X (gfx1250) — compile-verified, matrix operand order checked by reading, not run on hardware
//
#include <hip/hip_runtime.h>

#define NN 4096      // nodes
#define HH 4         // heads

typedef __attribute__((ext_vector_type(16))) __bf16       v16bf;
typedef __attribute__((ext_vector_type(8)))  __bf16       v8bf;
typedef __attribute__((ext_vector_type(8)))  float        v8f;
typedef __attribute__((ext_vector_type(8)))  unsigned int v8u;

// ---------- helpers ----------
static __device__ __forceinline__ float bf2f(unsigned short h) {
    return __builtin_bit_cast(float, ((unsigned)h) << 16);
}
static __device__ __forceinline__ v8f wmma_bf16(v16bf a, v16bf b, v8f c) {
    return __builtin_amdgcn_wmma_f32_16x16x32_bf16(
        /*neg_a=*/false, a, /*neg_b=*/false, b,
        /*c_mod=*/(short)0, c, /*reuse_a=*/false, /*reuse_b=*/false);
}
static __device__ __forceinline__ v16bf load_bf16_pair(const unsigned short* p0,
                                                       const unsigned short* p1) {
    uint4 b0 = *(const uint4*)p0;
    uint4 b1 = *(const uint4*)p1;
    v8u b;
    b[0] = b0.x; b[1] = b0.y; b[2] = b0.z; b[3] = b0.w;
    b[4] = b1.x; b[5] = b1.y; b[6] = b1.z; b[7] = b1.w;
    return __builtin_bit_cast(v16bf, b);
}

// ---------- kernel 1: W [H,Fin,Fout] f32 -> WT [H,Fout,Fin] bf16 ----------
__global__ void k_transposeW(const float* __restrict__ W,
                             unsigned short* __restrict__ WT,
                             int Fin, int Fout) {
    int idx = blockIdx.x * 256 + threadIdx.x;
    int tot = HH * Fin * Fout;
    if (idx >= tot) return;
    int i = idx % Fin;                 // fastest -> coalesced WT writes
    int r = idx / Fin;
    int o = r % Fout;
    int h = r / Fout;
    float v = W[((long)(h * Fin) + i) * Fout + o];
    WT[((long)(h * Fout) + o) * (long)Fin + i] =
        __builtin_bit_cast(unsigned short, (__bf16)v);
}

// ---------- kernel 2: hT[h][o][n] = (x @ W[h])^T  (bf16 out, WMMA) ----------
// 1 wave per block; wave computes 16 rows x (OTILES*16) cols.
template <int OTILES>
__global__ void k_gemm(const float* __restrict__ X,
                       const unsigned short* __restrict__ WT,
                       unsigned short* __restrict__ hT,
                       int Fin, int Fout) {
    const int lane = threadIdx.x;                 // 0..31
    const int rb   = blockIdx.x;                  // 16-row block
    const int ob   = blockIdx.y;                  // output band
    const int head = blockIdx.z;
    const int row  = rb * 16 + (lane & 15);
    const int kb   = (lane < 16) ? 0 : 8;

    v8f c[OTILES] = {};
    const float* xrow = X + (long)row * Fin;

    for (int k = 0; k < Fin; k += 32) {
        // A tile: x rows, native bf16 converts, CDNA5 16-bit A layout
        float4 x0 = *(const float4*)(xrow + k + kb);
        float4 x1 = *(const float4*)(xrow + k + kb + 4);
        float4 x2 = *(const float4*)(xrow + k + 16 + kb);
        float4 x3 = *(const float4*)(xrow + k + 20 + kb);
        v16bf a;
        a[0]  = (__bf16)x0.x; a[1]  = (__bf16)x0.y;
        a[2]  = (__bf16)x0.z; a[3]  = (__bf16)x0.w;
        a[4]  = (__bf16)x1.x; a[5]  = (__bf16)x1.y;
        a[6]  = (__bf16)x1.z; a[7]  = (__bf16)x1.w;
        a[8]  = (__bf16)x2.x; a[9]  = (__bf16)x2.y;
        a[10] = (__bf16)x2.z; a[11] = (__bf16)x2.w;
        a[12] = (__bf16)x3.x; a[13] = (__bf16)x3.y;
        a[14] = (__bf16)x3.z; a[15] = (__bf16)x3.w;
#pragma unroll
        for (int t = 0; t < OTILES; t++) {
            int o = ob * OTILES * 16 + t * 16 + (lane & 15);
            const unsigned short* wr = WT + ((long)(head * Fout) + o) * (long)Fin + k;
            v16bf b = load_bf16_pair(wr + kb, wr + 16 + kb);
            c[t] = wmma_bf16(a, b, c[t]);
        }
    }
    // store transposed: consecutive C VGPRs -> consecutive n -> one b128 store
    const int nbase = rb * 16 + ((lane < 16) ? 0 : 8);
#pragma unroll
    for (int t = 0; t < OTILES; t++) {
        int o = ob * OTILES * 16 + t * 16 + (lane & 15);
        v8bf s;
#pragma unroll
        for (int r = 0; r < 8; r++) s[r] = (__bf16)c[t][r];
        *(uint4*)(hT + ((long)(head * Fout) + o) * (long)NN + nbase) =
            __builtin_bit_cast(uint4, s);
    }
}

// ---------- kernel 3: f_src/f_dst [H,N] from hT and a [H,2,Fout] ----------
__global__ void k_f(const unsigned short* __restrict__ hT,
                    const float* __restrict__ a,
                    float* __restrict__ fs, float* __restrict__ fd, int Fout) {
    int n    = blockIdx.x * 32 + threadIdx.x;
    int head = blockIdx.z;
    const unsigned short* base = hT + (long)head * Fout * NN + n;
    const float* a0 = a + head * 2 * Fout;
    const float* a1 = a0 + Fout;
    float s0 = 0.f, s1 = 0.f;
    for (int o = 0; o < Fout; o++) {
        float v = bf2f(base[(long)o * NN]);
        s0 += v * a0[o];
        s1 += v * a1[o];
    }
    fs[head * NN + n] = s0;
    fd[head * NN + n] = s1;
}

// ---------- kernel 4: online masked-softmax stats per (head,row) ----------
__global__ void k_stats(const float* __restrict__ adj,
                        const float* __restrict__ fs, const float* __restrict__ fd,
                        float* __restrict__ mx, float* __restrict__ sm) {
    int row = blockIdx.x, head = blockIdx.z, lane = threadIdx.x;
    float fsr = fs[head * NN + row];
    const float* arow = adj + (long)row * NN;
    const float* fdh  = fd + head * NN;
    float m = -3.0e38f, s = 0.f;
    for (int c = lane; c < NN; c += 32) {
        if (arow[c] > 0.f) {
            float v = fsr + fdh[c];
            v = v > 0.f ? v : 0.2f * v;          // leaky_relu 0.2
            if (v > m) { s = s * __expf(m - v) + 1.f; m = v; }
            else       { s += __expf(v - m); }
        }
    }
    for (int off = 16; off > 0; off >>= 1) {      // wave32 reduce
        float m2 = __shfl_xor(m, off, 32);
        float s2 = __shfl_xor(s, off, 32);
        float M  = fmaxf(m, m2);
        s = s * __expf(m - M) + s2 * __expf(m2 - M);
        m = M;
    }
    if (lane == 0) { mx[head * NN + row] = m; sm[head * NN + row] = s; }
}

static __device__ __forceinline__ float attval(float av, float fdv,
                                               float fsr, float m, float rs) {
    float v = fsr + fdv;
    v = v > 0.f ? v : 0.2f * v;
    float e = __expf(v - m) * rs;
    return av > 0.f ? e : 0.f;
}

// ---------- kernel 5: out = elu(att @ h), att generated in registers ----------
template <int OTILES, bool FINAL>
__global__ void k_attn(const float* __restrict__ adj,
                       const unsigned short* __restrict__ hT,
                       const float* __restrict__ fs, const float* __restrict__ fd,
                       const float* __restrict__ mx, const float* __restrict__ sm,
                       float* __restrict__ out, int Fout, int outW) {
    const int lane = threadIdx.x;
    const int rb = blockIdx.x, ob = blockIdx.y, head = blockIdx.z;
    const int row = rb * 16 + (lane & 15);
    const int kb  = (lane < 16) ? 0 : 8;

    const float fsr = fs[head * NN + row];
    const float m   = mx[head * NN + row];
    const float rs  = 1.f / sm[head * NN + row];

    v8f c[OTILES] = {};
    const float* arow = adj + (long)row * NN;
    const float* fdh  = fd + head * NN;

    for (int k = 0; k < NN; k += 32) {
        float4 a0 = *(const float4*)(arow + k + kb);
        float4 a1 = *(const float4*)(arow + k + kb + 4);
        float4 a2 = *(const float4*)(arow + k + 16 + kb);
        float4 a3 = *(const float4*)(arow + k + 20 + kb);
        float4 d0 = *(const float4*)(fdh + k + kb);
        float4 d1 = *(const float4*)(fdh + k + kb + 4);
        float4 d2 = *(const float4*)(fdh + k + 16 + kb);
        float4 d3 = *(const float4*)(fdh + k + 20 + kb);
        float av[16] = {a0.x,a0.y,a0.z,a0.w, a1.x,a1.y,a1.z,a1.w,
                        a2.x,a2.y,a2.z,a2.w, a3.x,a3.y,a3.z,a3.w};
        float dv[16] = {d0.x,d0.y,d0.z,d0.w, d1.x,d1.y,d1.z,d1.w,
                        d2.x,d2.y,d2.z,d2.w, d3.x,d3.y,d3.z,d3.w};
        v16bf A;
#pragma unroll
        for (int j = 0; j < 16; j++)
            A[j] = (__bf16)attval(av[j], dv[j], fsr, m, rs);
#pragma unroll
        for (int t = 0; t < OTILES; t++) {
            int o = ob * OTILES * 16 + t * 16 + (lane & 15);
            const unsigned short* hr = hT + ((long)(head * Fout) + o) * (long)NN + k;
            v16bf B = load_bf16_pair(hr + kb, hr + 16 + kb);
            c[t] = wmma_bf16(A, B, c[t]);
        }
    }
    // ELU (+ extra final ELU), head-concatenated store
    const int nbase = rb * 16 + ((lane < 16) ? 0 : 8);
#pragma unroll
    for (int t = 0; t < OTILES; t++) {
        int o = head * Fout + ob * OTILES * 16 + t * 16 + (lane & 15);
#pragma unroll
        for (int r = 0; r < 8; r++) {
            float v = c[t][r];
            v = v > 0.f ? v : __expf(v) - 1.f;
            if (FINAL) v = v > 0.f ? v : __expf(v) - 1.f;
            out[(long)(nbase + r) * outW + o] = v;
        }
    }
}

// ---------- host driver ----------
extern "C" void kernel_launch(void* const* d_in, const int* in_sizes, int n_in,
                              void* d_out, int out_size, void* d_ws, size_t ws_size,
                              hipStream_t stream) {
    const float* node = (const float*)d_in[0];
    const float* adj  = (const float*)d_in[3];
    const float* Wl[3] = {(const float*)d_in[4], (const float*)d_in[6], (const float*)d_in[8]};
    const float* al[3] = {(const float*)d_in[5], (const float*)d_in[7], (const float*)d_in[9]};

    char* ws = (char*)d_ws;
    const size_t MB = 1024 * 1024;
    float*          x0 = (float*)(ws);                       // 32 MB  [N,2048]
    float*          x1 = (float*)(ws + 32 * MB);             // 32 MB  [N,2048]
    unsigned short* hT = (unsigned short*)(ws + 64 * MB);    // 16 MB  [H,512,N] bf16
    unsigned short* WT = (unsigned short*)(ws + 80 * MB);    //  8 MB  [H,Fout,Fin] bf16
    float*          fs = (float*)(ws + 88 * MB);             // [H,N]
    float*          fd = fs + HH * NN;
    float*          mx = fd + HH * NN;
    float*          sm = mx + HH * NN;

    const int   Fin[3]  = {512, 2048, 2048};
    const int   Fout[3] = {512, 512, 64};
    const float* xin[3] = {node, x0, x1};
    float*      xout[2] = {x0, x1};

    for (int l = 0; l < 3; l++) {
        int fin = Fin[l], fout = Fout[l];
        int tot = HH * fin * fout;
        k_transposeW<<<(tot + 255) / 256, 256, 0, stream>>>(Wl[l], WT, fin, fout);

        dim3 gg(NN / 16, fout / 64, HH);
        k_gemm<4><<<gg, 32, 0, stream>>>(xin[l], WT, hT, fin, fout);

        k_f<<<dim3(NN / 32, 1, HH), 32, 0, stream>>>(hT, al[l], fs, fd, fout);
        k_stats<<<dim3(NN, 1, HH), 32, 0, stream>>>(adj, fs, fd, mx, sm);

        if (l < 2) {
            k_attn<8, false><<<dim3(NN / 16, fout / 128, HH), 32, 0, stream>>>(
                adj, hT, fs, fd, mx, sm, xout[l], fout, HH * fout);
        } else {
            k_attn<4, true><<<dim3(NN / 16, 1, HH), 32, 0, stream>>>(
                adj, hT, fs, fd, mx, sm, (float*)d_out, fout, HH * fout);
        }
    }
}
